// MultiHeadAttention_6150393168354
// MI455X (gfx1250) — matrix operand order repaired and checked by reading, not compiled, not run on hardware
//
#include <hip/hip_runtime.h>
#include <stdint.h>

#define BATCH 2
#define SEQ   2048
#define HIDN  2048
#define NH    16
#define HD    128

typedef __attribute__((ext_vector_type(16))) __bf16 v16bf;
typedef __attribute__((ext_vector_type(8)))  float  v8f;
typedef __attribute__((ext_vector_type(2)))  float  v2f;
typedef __attribute__((ext_vector_type(2)))  __bf16 v2bf;

union FragBF { v16bf v; uint32_t u[8]; };
union FragF  { v8f   v; float    f[8]; };

// f32 -> bf16 (RNE), native v_cvt on gfx1250.
__device__ __forceinline__ unsigned short f2bf(float x) {
  __bf16 h = (__bf16)x;
  return __builtin_bit_cast(unsigned short, h);
}
// Packed pair -> single v_cvt_pk_bf16_f32, no repack.
__device__ __forceinline__ uint32_t f2bf2(float lo, float hi) {
  v2f x; x.x = lo; x.y = hi;
  v2bf h = __builtin_convertvector(x, v2bf);
  return __builtin_bit_cast(uint32_t, h);
}

// A-operand fragment (16x32 bf16). rowbase = u32 view of LDS row (stride 18 u32).
// Per ISA: lanes 0-15 hold K {0..7,16..23}; lanes 16-31 hold K {8..15,24..31}.
__device__ __forceinline__ void loadFragA(FragBF& f, const uint32_t* rowbase, int half) {
  const int o = half * 4;
#pragma unroll
  for (int j = 0; j < 4; ++j) { f.u[j] = rowbase[j + o]; f.u[j + 4] = rowbase[8 + j + o]; }
}

// B-operand fragment (32x16 bf16). lanes 0-15: K {0..7,16..23}; lanes 16-31: K {8..15,24..31}.
__device__ __forceinline__ void loadFragB(FragBF& f, const uint32_t* rowbase, int half) {
  const int o = half * 4;
#pragma unroll
  for (int j = 0; j < 4; ++j) { f.u[j] = rowbase[j + o]; f.u[j + 4] = rowbase[8 + j + o]; }
}

__device__ __forceinline__ v8f wmma_bf16(const FragBF& a, const FragBF& b, v8f c) {
  return __builtin_amdgcn_wmma_f32_16x16x32_bf16(false, a.v, false, b.v, (short)0, c,
                                                 false, false);
}

// ---------------------------------------------------------------------------
// Generic bf16-WMMA GEMM: C[M,N] = A[M,K] @ B[K,N] (fp32 in/out, bf16 math).
// Block tile 128x128, K-step 32, 8 waves (2x4), wave tile 64x32 (4x2 wmma tiles).
// mode 0: plain store to C.  mode 1: scatter qkv columns into q/k/v [B][H][S][D].
// ---------------------------------------------------------------------------
#define BM 128
#define BN 128
#define BK 32
#define LDSS 36      // padded bf16 stride (even -> u32 aligned pairs)

__global__ __launch_bounds__(256) void gemm_bf16(
    const float* __restrict__ A, const float* __restrict__ B, float* __restrict__ C,
    int M, int N, int K, float* qb, float* kb_, float* vb, int mode)
{
  __shared__ unsigned short aLds[BM * LDSS];
  __shared__ unsigned short bLds[BN * LDSS];
  const int tid  = threadIdx.x;
  const int lane = tid & 31;
  const int wave = tid >> 5;
  const int waveM = wave >> 2;      // 0..1
  const int waveN = wave & 3;       // 0..3
  const int lm   = lane & 15;
  const int half = lane >> 4;
  const int bm   = blockIdx.y * BM;
  const int bn   = blockIdx.x * BN;

  FragF acc[4][2];
#pragma unroll
  for (int i = 0; i < 4; ++i)
#pragma unroll
    for (int j = 0; j < 2; ++j)
#pragma unroll
      for (int e = 0; e < 8; ++e) acc[i][j].f[e] = 0.0f;

  const uint32_t* aLds32 = (const uint32_t*)aLds;
  const uint32_t* bLds32 = (const uint32_t*)bLds;

  for (int k0 = 0; k0 < K; k0 += BK) {
    // Stage A tile (BMxBK) as bf16: fixed-trip, float4 loads (coalesced along k).
#pragma unroll
    for (int it = 0; it < (BM * BK / 4) / 256; ++it) {
      const int i = tid + it * 256;
      const int r = i >> 3, c4 = (i & 7) * 4;
      const float4 a = *(const float4*)&A[(size_t)(bm + r) * K + k0 + c4];
      uint32_t* dst = (uint32_t*)&aLds[r * LDSS + c4];
      dst[0] = f2bf2(a.x, a.y);
      dst[1] = f2bf2(a.z, a.w);
    }
    // Stage B tile transposed (BNxBK): fixed-trip, float4 loads (coalesced along n).
#pragma unroll
    for (int it = 0; it < (BK * BN / 4) / 256; ++it) {
      const int i = tid + it * 256;
      const int r = i >> 5, c4 = (i & 31) * 4;        // r = k row, c4 = n col
      const float4 bv = *(const float4*)&B[(size_t)(k0 + r) * N + bn + c4];
      bLds[(c4 + 0) * LDSS + r] = f2bf(bv.x);
      bLds[(c4 + 1) * LDSS + r] = f2bf(bv.y);
      bLds[(c4 + 2) * LDSS + r] = f2bf(bv.z);
      bLds[(c4 + 3) * LDSS + r] = f2bf(bv.w);
    }
    __syncthreads();

    FragBF af[4], bf_[2];
#pragma unroll
    for (int i = 0; i < 4; ++i)
      loadFragA(af[i], aLds32 + (waveM * 64 + i * 16 + lm) * (LDSS / 2), half);
#pragma unroll
    for (int j = 0; j < 2; ++j)
      loadFragB(bf_[j], bLds32 + (waveN * 32 + j * 16 + lm) * (LDSS / 2), half);

#pragma unroll
    for (int i = 0; i < 4; ++i)
#pragma unroll
      for (int j = 0; j < 2; ++j)
        acc[i][j].v = wmma_bf16(af[i], bf_[j], acc[i][j].v);
    __syncthreads();
  }

  // Epilogue
#pragma unroll
  for (int i = 0; i < 4; ++i) {
#pragma unroll
    for (int j = 0; j < 2; ++j) {
#pragma unroll
      for (int e = 0; e < 8; ++e) {
        const int row = bm + waveM * 64 + i * 16 + e + 8 * half;   // C layout: M = e + 8*half
        const int col = bn + waveN * 32 + j * 16 + lm;             // N = lane&15
        const float val = acc[i][j].f[e];
        if (mode == 0) {
          C[(size_t)row * N + col] = val;
        } else {
          const int which = col >> 11;          // 0=q 1=k 2=v (3*HID columns)
          const int rem   = col & 2047;
          const int h     = rem >> 7;
          const int d     = rem & 127;
          const int b     = row >> 11;          // M = B*S, S = 2048
          const int s     = row & 2047;
          float* dst = (which == 0) ? qb : (which == 1) ? kb_ : vb;
          dst[(((size_t)b * NH + h) * SEQ + s) * HD + d] = val;
        }
      }
    }
  }
}

// ---------------------------------------------------------------------------
// RoPE (in-place on q and k, layout [B][H][S][D]); one thread per (bh,s,d<64).
// ---------------------------------------------------------------------------
__global__ __launch_bounds__(256) void rope_kernel(float* __restrict__ q,
                                                   float* __restrict__ k)
{
  const int idx = blockIdx.x * blockDim.x + threadIdx.x;
  const int d  = idx & 63;
  const int s  = (idx >> 6) & (SEQ - 1);
  const int bh = idx >> 17;
  if (bh >= BATCH * NH) return;
  const float inv = __powf(10000.0f, -(float)(2 * d) / 128.0f);
  float sn, cs;
  __sincosf((float)s * inv, &sn, &cs);
  const size_t base = ((size_t)bh * SEQ + s) * HD;
  float a0 = q[base + d], a1 = q[base + d + 64];
  q[base + d]      = a0 * cs - a1 * sn;
  q[base + d + 64] = a1 * cs + a0 * sn;
  float b0 = k[base + d], b1 = k[base + d + 64];
  k[base + d]      = b0 * cs - b1 * sn;
  k[base + d + 64] = b1 * cs + b0 * sn;
}

// ---------------------------------------------------------------------------
// Causal flash attention. Block = 4 waves = 64 q rows of one (b,h).
// Streams K/V in 32-row chunks; Q in registers; online softmax; bf16 WMMA.
// Output written as [B][S][H*D] (fp32) to feed the output projection GEMM.
// ---------------------------------------------------------------------------
#define QB 64
#define KC 32
#define KLS 132      // kLds row stride in bf16 (rows = k, cols = d[128])

__global__ __launch_bounds__(128) void attn_kernel(
    const float* __restrict__ Q, const float* __restrict__ K,
    const float* __restrict__ V, float* __restrict__ Out)
{
  __shared__ unsigned short kLds[KC * KLS];        // [krow][d]
  __shared__ unsigned short vLds[HD * LDSS];       // [d][krow]  (V transposed)
  __shared__ unsigned short pLds[4][16 * LDSS];    // per-wave P tile [16][32]

  const int tid  = threadIdx.x;
  const int lane = tid & 31;
  const int wave = tid >> 5;
  const int lm   = lane & 15;
  const int half = lane >> 4;
  const int bh   = blockIdx.y;
  const int b    = bh >> 4;
  const int h    = bh & 15;
  const int qblk = blockIdx.x * QB;
  const int qbase = qblk + wave * 16;

  const float* Qp = Q + (size_t)bh * SEQ * HD;
  const float* Kp = K + (size_t)bh * SEQ * HD;
  const float* Vp = V + (size_t)bh * SEQ * HD;

  // Q fragments (A-operand): 4 K-steps of 32 over D=128; pairs are contiguous.
  FragBF qf[4];
  {
    const float* qrow = Qp + (size_t)(qbase + lm) * HD;
#pragma unroll
    for (int ds = 0; ds < 4; ++ds) {
#pragma unroll
      for (int j = 0; j < 8; ++j) {
        const int kk = ds * 32 + ((j < 4) ? 2 * j : 16 + 2 * (j - 4)) + half * 8;
        const float2 p = *(const float2*)&qrow[kk];
        qf[ds].u[j] = f2bf2(p.x, p.y);
      }
    }
  }

  FragF o[8];
#pragma unroll
  for (int dt = 0; dt < 8; ++dt)
#pragma unroll
    for (int e = 0; e < 8; ++e) o[dt].f[e] = 0.0f;
  float mrow[8], lrow[8];
#pragma unroll
  for (int e = 0; e < 8; ++e) { mrow[e] = -1e30f; lrow[e] = 0.0f; }
  const float scale = 0.08838834764831845f;   // 1/sqrt(128)

  const uint32_t* kLds32 = (const uint32_t*)kLds;
  const uint32_t* vLds32 = (const uint32_t*)vLds;

  const int kend = qblk + QB;
  for (int kb = 0; kb < kend; kb += KC) {
    // Stage K chunk [KC][HD] and V chunk transposed [HD][KC]: fixed-trip float4.
#pragma unroll
    for (int it = 0; it < (KC * HD / 4) / 128; ++it) {
      const int i = tid + it * 128;
      const int r = i >> 5, c4 = (i & 31) * 4;
      const float4 kv = *(const float4*)&Kp[(size_t)(kb + r) * HD + c4];
      const float4 vv = *(const float4*)&Vp[(size_t)(kb + r) * HD + c4];
      uint32_t* kd = (uint32_t*)&kLds[r * KLS + c4];
      kd[0] = f2bf2(kv.x, kv.y);
      kd[1] = f2bf2(kv.z, kv.w);
      vLds[(c4 + 0) * LDSS + r] = f2bf(vv.x);
      vLds[(c4 + 1) * LDSS + r] = f2bf(vv.y);
      vLds[(c4 + 2) * LDSS + r] = f2bf(vv.z);
      vLds[(c4 + 3) * LDSS + r] = f2bf(vv.w);
    }
    __syncthreads();

    // Scores: two 16x16 tiles (k cols kb..kb+31), contraction over D=128.
    FragF sc[2];
#pragma unroll
    for (int t = 0; t < 2; ++t) {
#pragma unroll
      for (int e = 0; e < 8; ++e) sc[t].f[e] = 0.0f;
      const uint32_t* rowb = kLds32 + (t * 16 + lm) * (KLS / 2);
#pragma unroll
      for (int ds = 0; ds < 4; ++ds) {
        FragBF kf;
        loadFragB(kf, rowb + ds * 16, half);
        sc[t].v = wmma_bf16(qf[ds], kf, sc[t].v);
      }
    }

    // Mask + online softmax (row = e + 8*half, spread over one 16-lane half).
    float alpha[8];
#pragma unroll
    for (int e = 0; e < 8; ++e) {
      const int qg = qbase + e + 8 * half;
      float s0 = sc[0].f[e] * scale;
      float s1 = sc[1].f[e] * scale;
      if (kb + lm      > qg) s0 = -1e30f;
      if (kb + 16 + lm > qg) s1 = -1e30f;
      float cm = fmaxf(s0, s1);
#pragma unroll
      for (int m = 1; m < 16; m <<= 1) cm = fmaxf(cm, __shfl_xor(cm, m, 32));
      const float nm = fmaxf(mrow[e], cm);
      const float al = __expf(mrow[e] - nm);
      const float p0 = __expf(s0 - nm);
      const float p1 = __expf(s1 - nm);
      float rs = p0 + p1;
#pragma unroll
      for (int m = 1; m < 16; m <<= 1) rs += __shfl_xor(rs, m, 32);
      lrow[e] = lrow[e] * al + rs;
      mrow[e] = nm;
      alpha[e] = al;
      sc[0].f[e] = p0;
      sc[1].f[e] = p1;
    }
#pragma unroll
    for (int dt = 0; dt < 8; ++dt)
#pragma unroll
      for (int e = 0; e < 8; ++e) o[dt].f[e] *= alpha[e];

    // P (C layout) -> per-wave LDS -> A layout for P @ V.
    unsigned short* pw = pLds[wave];
#pragma unroll
    for (int e = 0; e < 8; ++e) {
      const int r = e + 8 * half;
      pw[r * LDSS + lm]      = f2bf(sc[0].f[e]);
      pw[r * LDSS + 16 + lm] = f2bf(sc[1].f[e]);
    }
    asm volatile("s_wait_dscnt 0" ::: "memory");

    FragBF pf;
    loadFragA(pf, (const uint32_t*)pw + lm * (LDSS / 2), half);
#pragma unroll
    for (int dt = 0; dt < 8; ++dt) {
      FragBF vf;
      loadFragB(vf, vLds32 + (dt * 16 + lm) * (LDSS / 2), half);
      o[dt].v = wmma_bf16(pf, vf, o[dt].v);
    }
    __syncthreads();
  }

  // Normalize and write [B][S][H*D].
#pragma unroll
  for (int e = 0; e < 8; ++e) lrow[e] = 1.0f / lrow[e];
#pragma unroll
  for (int dt = 0; dt < 8; ++dt) {
#pragma unroll
    for (int e = 0; e < 8; ++e) {
      const int s = qbase + e + 8 * half;
      const int d = dt * 16 + lm;
      Out[((size_t)b * SEQ + s) * HIDN + h * HD + d] = o[dt].f[e] * lrow[e];
    }
  }
}

// ---------------------------------------------------------------------------
extern "C" void kernel_launch(void* const* d_in, const int* in_sizes, int n_in,
                              void* d_out, int out_size, void* d_ws, size_t ws_size,
                              hipStream_t stream) {
  const float* x     = (const float*)d_in[0];
  const float* w_qkv = (const float*)d_in[1];
  const float* w_out = (const float*)d_in[2];
  float* out = (float*)d_out;

  const size_t perBuf = (size_t)BATCH * NH * SEQ * HD;   // 8M floats = 32MB
  float* qB = (float*)d_ws;
  float* kB = qB + perBuf;
  float* vB = kB + perBuf;
  float* aB = vB + perBuf;

  // 1) QKV projection (scatter epilogue into q/k/v [B][H][S][D])
  {
    dim3 grid((3 * HIDN) / BN, (BATCH * SEQ) / BM);
    gemm_bf16<<<grid, 256, 0, stream>>>(x, w_qkv, nullptr,
                                        BATCH * SEQ, 3 * HIDN, HIDN,
                                        qB, kB, vB, 1);
  }
  // 2) RoPE on q, k
  {
    const int total = BATCH * NH * SEQ * 64;
    rope_kernel<<<total / 256, 256, 0, stream>>>(qB, kB);
  }
  // 3) Causal flash attention -> aB [B][S][H*D]
  {
    dim3 grid(SEQ / QB, BATCH * NH);
    attn_kernel<<<grid, 128, 0, stream>>>(qB, kB, vB, aB);
  }
  // 4) Output projection
  {
    dim3 grid(HIDN / BN, (BATCH * SEQ) / BM);
    gemm_bf16<<<grid, 256, 0, stream>>>(aB, w_out, out,
                                        BATCH * SEQ, HIDN, HIDN,
                                        nullptr, nullptr, nullptr, 0);
  }
}
